// MoERouter_37933151158615
// MI455X (gfx1250) — matrix operand order repaired and checked by reading, not compiled, not run on hardware
//
#include <hip/hip_runtime.h>
#include <hip/hip_bf16.h>

#define T_TOK   16384
#define HDIM    7168
#define NEXP    256
#define NGROUP  8
#define EPG     32      // experts per group
#define TOPKG   4
#define TOPK    8
#define MT      4       // M (token) tiles per wave
#define MBLK    64      // tokens per block (MT * 16)

typedef __attribute__((ext_vector_type(16))) __bf16 bf16x16;
typedef __attribute__((ext_vector_type(16))) float  f32x16;
typedef __attribute__((ext_vector_type(8)))  float  f32x8;

// 16 f32 -> 16 bf16 via native packed converts (v_cvt_pk_bf16_f32).
__device__ __forceinline__ bf16x16 to_bf16x16(float4 a, float4 b, float4 c, float4 d) {
    f32x16 f;
    f[0]  = a.x; f[1]  = a.y; f[2]  = a.z; f[3]  = a.w;
    f[4]  = b.x; f[5]  = b.y; f[6]  = b.z; f[7]  = b.w;
    f[8]  = c.x; f[9]  = c.y; f[10] = c.z; f[11] = c.w;
    f[12] = d.x; f[13] = d.y; f[14] = d.z; f[15] = d.w;
    return __builtin_convertvector(f, bf16x16);
}

// Gate GEMM: logits[T,E] = X[T,H] * W[E,H]^T via bf16 WMMA, f32 accumulate.
// Block: 256 threads = 8 waves, covering 64 tokens x 256 experts.
// Wave w: experts [32w, 32w+32) as two 16x16 N-tiles, for 4 M-tiles.
// Per K-step(32): 8 WMMAs; A reused across N-tiles, B reused across M-tiles.
__global__ __launch_bounds__(256) void gate_gemm_kernel(
    const float* __restrict__ X, const float* __restrict__ W,
    float* __restrict__ logits)
{
    const int lane = threadIdx.x & 31;
    const int wave = threadIdx.x >> 5;
    const int l15  = lane & 15;
    const int half = lane >> 4;            // 0: lanes 0-15, 1: lanes 16-31
    const int m0   = blockIdx.x * MBLK;    // token tile base
    const int n0   = wave * 32;            // expert base for this wave

    // A operand (contiguous K order, same as B): lanes 0-15 hold K=0..15 of
    // row M=l15; lanes 16-31 hold K=16..31.
    size_t xbase[MT];
#pragma unroll
    for (int mt = 0; mt < MT; ++mt)
        xbase[mt] = (size_t)(m0 + mt * 16 + l15) * HDIM + (half ? 16 : 0);

    // B operand (contiguous K order, same as A): lane holds 16 consecutive K of
    // column N=l15 (lanes 0-15: K 0..15, lanes 16-31: K 16..31).
    // B[k][n] = W[n][k] -> contiguous in W's row.
    const size_t wbase[2] = {
        (size_t)(n0      + l15) * HDIM + (half ? 16 : 0),
        (size_t)(n0 + 16 + l15) * HDIM + (half ? 16 : 0)
    };

    f32x8 acc[MT][2];
#pragma unroll
    for (int mt = 0; mt < MT; ++mt)
#pragma unroll
        for (int nt = 0; nt < 2; ++nt)
            acc[mt][nt] = 0.0f;

    for (int k0 = 0; k0 < HDIM; k0 += 32) {
        // ---- A fragments: 16 consecutive K from xbase ----
        bf16x16 a[MT];
#pragma unroll
        for (int mt = 0; mt < MT; ++mt) {
            const float4* p = (const float4*)(X + xbase[mt] + k0);
            a[mt] = to_bf16x16(p[0], p[1], p[2], p[3]);
        }

        // ---- B fragments + WMMAs; B reused across all 4 M tiles ----
#pragma unroll
        for (int nt = 0; nt < 2; ++nt) {
            const float4* q = (const float4*)(W + wbase[nt] + k0);
            bf16x16 b = to_bf16x16(q[0], q[1], q[2], q[3]);
#pragma unroll
            for (int mt = 0; mt < MT; ++mt)
                acc[mt][nt] = __builtin_amdgcn_wmma_f32_16x16x32_bf16(
                    false, a[mt], false, b, (short)0, acc[mt][nt], false, false);
        }
    }

    // C/D layout: VGPR r, lanes 0-15 -> M=r, lanes 16-31 -> M=r+8; N = l15.
#pragma unroll
    for (int mt = 0; mt < MT; ++mt) {
#pragma unroll
        for (int r = 0; r < 8; ++r) {
            const int orow = m0 + mt * 16 + r + half * 8;
            logits[(size_t)orow * NEXP + n0 +      l15] = acc[mt][0][r];
            logits[(size_t)orow * NEXP + n0 + 16 + l15] = acc[mt][1][r];
        }
    }
}

__device__ __forceinline__ float sigmoidf(float x) {
    return 1.0f / (1.0f + __expf(-x));
}

// One thread per token: group sums -> top-4 groups -> masked top-8 experts
// -> normalized weights + int64 indices.
__global__ __launch_bounds__(256) void router_kernel(
    const float* __restrict__ logits,
    float* __restrict__ topw, long long* __restrict__ topi)
{
    const int t = blockIdx.x * blockDim.x + threadIdx.x;
    if (t >= T_TOK) return;
    const float* lrow = logits + (size_t)t * NEXP;

    // Pass 1: per-group sigmoid sums.
    float gs[NGROUP];
#pragma unroll
    for (int g = 0; g < NGROUP; ++g) {
        float s = 0.0f;
        for (int j = 0; j < EPG; ++j)
            s += sigmoidf(lrow[g * EPG + j]);
        gs[g] = s;
    }

    // Top-4 groups, first-index tie-break (matches lax.top_k).
    unsigned sel = 0;
#pragma unroll
    for (int it = 0; it < TOPKG; ++it) {
        float best = -1.0f; int bg = 0;
#pragma unroll
        for (int g = 0; g < NGROUP; ++g) {
            const bool avail = ((sel >> g) & 1u) == 0u;
            if (avail && gs[g] > best) { best = gs[g]; bg = g; }
        }
        sel |= (1u << bg);
    }

    // Pass 2: masked top-8 experts via unrolled 8-slot insertion.
    float     w[TOPK];
    long long ix[TOPK];
#pragma unroll
    for (int r = 0; r < TOPK; ++r) { w[r] = -1.0f; ix[r] = 0; }

    for (int e = 0; e < NEXP; ++e) {
        const int g = e >> 5;
        float s = sigmoidf(lrow[e]);
        s = ((sel >> g) & 1u) ? s : 0.0f;
        float cv = s; long long ci = (long long)e;
#pragma unroll
        for (int r = 0; r < TOPK; ++r) {
            if (cv > w[r]) {
                const float tw = w[r]; const long long ti = ix[r];
                w[r] = cv; ix[r] = ci;
                cv = tw; ci = ti;
            }
        }
    }

    float denom = 0.0f;
#pragma unroll
    for (int r = 0; r < TOPK; ++r) denom += w[r];
    denom = fmaxf(denom, 1e-12f);
    const float inv = 1.0f / denom;
#pragma unroll
    for (int r = 0; r < TOPK; ++r) {
        topw[(size_t)t * TOPK + r] = w[r] * inv;
        topi[(size_t)t * TOPK + r] = ix[r];
    }
}

extern "C" void kernel_launch(void* const* d_in, const int* in_sizes, int n_in,
                              void* d_out, int out_size, void* d_ws, size_t ws_size,
                              hipStream_t stream) {
    (void)in_sizes; (void)n_in; (void)out_size; (void)d_ws; (void)ws_size;
    const float* X = (const float*)d_in[0];   // [16384, 7168] f32
    const float* W = (const float*)d_in[1];   // [256, 7168] f32

    // Output layout (concatenated in reference return order):
    //   topk_weights: 16384*8 f32
    //   topk_indices: 16384*8 i64
    //   logits:       16384*256 f32
    float*     topw   = (float*)d_out;
    long long* topi   = (long long*)((char*)d_out + (size_t)T_TOK * TOPK * sizeof(float));
    float*     logits = (float*)((char*)d_out + (size_t)T_TOK * TOPK * (sizeof(float) + sizeof(long long)));

    gate_gemm_kernel<<<T_TOK / MBLK, 256, 0, stream>>>(X, W, logits);
    router_kernel<<<(T_TOK + 255) / 256, 256, 0, stream>>>(logits, topw, topi);
}
